// EncodedGINE_19301583028826
// MI455X (gfx1250) — matrix operand order repaired and checked by reading, not compiled, not run on hardware
//
#include <hip/hip_runtime.h>
#include <hip/hip_bf16.h>
#include <math.h>

typedef __attribute__((ext_vector_type(16))) __bf16 v16bf;
typedef __attribute__((ext_vector_type(8)))  __bf16 v8bf;
typedef __attribute__((ext_vector_type(8)))  float  v8f;

#define DIM 256
#define NGRAPH 4096
#define NTASK 12
#define MIDSTRIDE 264                      // 256 + 8 bf16 pad -> 4-dword bank skew per row
#define WLDS_ELEMS (DIM * DIM)             // 65536 bf16 = 128 KB staged weight
#define MID_ELEMS (8 * 16 * MIDSTRIDE)     // 33792 bf16 = 67.6 KB mid tiles
#define SMEM_BYTES ((WLDS_ELEMS + MID_ELEMS) * 2)

// ---------------------------------------------------------------------------
// Small prep kernels (tiny cost, run once per launch)
// ---------------------------------------------------------------------------

// node_tbl[r][c] = sum_k node_emb_i[v][k] * npw[(i*256+k)*256 + c]   (177 rows)
__global__ void node_table_kernel(const float* __restrict__ e0, const float* __restrict__ e1,
                                  const float* __restrict__ e2, const float* __restrict__ e3,
                                  const float* __restrict__ e4, const float* __restrict__ e5,
                                  const float* __restrict__ e6, const float* __restrict__ e7,
                                  const float* __restrict__ e8,
                                  const float* __restrict__ W, float* __restrict__ out) {
  const int off[10] = {0,119,128,139,151,160,165,173,175,177};
  const float* tbl[9] = {e0,e1,e2,e3,e4,e5,e6,e7,e8};
  int r = blockIdx.x, c = threadIdx.x;
  int i = 0;
  while (r >= off[i+1]) ++i;
  const float* emb = tbl[i] + (size_t)(r - off[i]) * DIM;
  const float* w = W + (size_t)(i * DIM) * DIM + c;
  float acc = 0.f;
  for (int k = 0; k < DIM; ++k) acc += emb[k] * w[(size_t)k * DIM];
  out[(size_t)r * DIM + c] = acc;
}

// edge base table (30 rows): ebase[r][c] = sum_k edge_emb_i[v][k]*epw[(i*256+k)*256+c]
__global__ void edge_base_kernel(const float* __restrict__ e0, const float* __restrict__ e1,
                                 const float* __restrict__ e2,
                                 const float* __restrict__ W, float* __restrict__ out) {
  const int off[4] = {0,22,28,30};
  const float* tbl[3] = {e0,e1,e2};
  int r = blockIdx.x, c = threadIdx.x;
  int i = 0;
  while (r >= off[i+1]) ++i;
  const float* emb = tbl[i] + (size_t)(r - off[i]) * DIM;
  const float* w = W + (size_t)(i * DIM) * DIM + c;
  float acc = 0.f;
  for (int k = 0; k < DIM; ++k) acc += emb[k] * w[(size_t)k * DIM];
  out[(size_t)r * DIM + c] = acc;
}

// per-layer folded edge table: rows 0..29 = ebase @ lin_w ; row 30 -> bias = epb@lin_w + lin_b
__global__ void edge_layer_kernel(const float* __restrict__ ebase, const float* __restrict__ epb,
                                  const float* __restrict__ linw, const float* __restrict__ linb,
                                  float* __restrict__ outTbl, float* __restrict__ outBias) {
  int r = blockIdx.x, c = threadIdx.x;
  const float* in = (r < 30) ? (ebase + (size_t)r * DIM) : epb;
  float acc = 0.f;
  for (int k = 0; k < DIM; ++k) acc += in[k] * linw[(size_t)k * DIM + c];
  if (r < 30) outTbl[(size_t)r * DIM + c] = acc;
  else        outBias[c] = acc + linb[c];
}

// Pack a 256x256 f32 weight (row-major [k][c]) into bf16 B-fragment order:
// idx = (((ks*16+nt)*32 + lane)*16 + j) ; k = 32*ks + 8*(lane>>4) + j + 8*(j>>3) ; c = 16*nt + (lane&15)
__global__ void pack_wfrag_kernel(const float* __restrict__ W, __bf16* __restrict__ out) {
  int idx  = blockIdx.x * 256 + threadIdx.x;     // 0..65535
  int j    = idx & 15;
  int lane = (idx >> 4) & 31;
  int nt   = (idx >> 9) & 15;
  int ks   = idx >> 13;
  int k = 32 * ks + ((lane >> 4) << 3) + j + ((j >> 3) << 3);
  int c = nt * 16 + (lane & 15);
  out[idx] = (__bf16)W[(size_t)k * DIM + c];
}

// fold BN(eval) + b1 into per-channel scale/shift:  y = (h@w1)*scale + shift
__global__ void bnfold_kernel(const float* __restrict__ b1, const float* __restrict__ g,
                              const float* __restrict__ b,  const float* __restrict__ m,
                              const float* __restrict__ v,
                              float* __restrict__ scale, float* __restrict__ shift) {
  int c = threadIdx.x;
  float S = g[c] * rsqrtf(v[c] + 1e-5f);
  scale[c] = S;
  shift[c] = (b1[c] - m[c]) * S + b[c];
}

// ---------------------------------------------------------------------------
// Node encoder: x[n] = sum_i node_tbl[off_i + feat_i] + node_proj_b
// ---------------------------------------------------------------------------
__global__ void node_encode_kernel(const int* __restrict__ xfeat, const float* __restrict__ ntbl,
                                   const float* __restrict__ npb, float* __restrict__ x) {
  const int off[9] = {0,119,128,139,151,160,165,173,175};
  int n = blockIdx.x, c = threadIdx.x;
  const int* f = xfeat + (size_t)n * 9;
  float acc = npb[c];
#pragma unroll
  for (int i = 0; i < 9; ++i) acc += ntbl[(size_t)(off[i] + f[i]) * DIM + c];
  x[(size_t)n * DIM + c] = acc;
}

// ---------------------------------------------------------------------------
// Edge message + scatter: h[dst] += relu(x[src] + folded_edge(e))   (h starts = x)
// one wave per edge; lane covers 8 channels via two float4s
// ---------------------------------------------------------------------------
__global__ void edge_msg_kernel(const float* __restrict__ x, const int* __restrict__ eidx,
                                const int* __restrict__ eattr, const float* __restrict__ etbl,
                                const float* __restrict__ ebias, float* __restrict__ h, int nE) {
  int wave = threadIdx.x >> 5;
  int lane = threadIdx.x & 31;
  int e = blockIdx.x * 8 + wave;
  if (e >= nE) return;
  int s  = eidx[e];
  int d  = eidx[nE + e];
  int f0 = eattr[(size_t)e * 3 + 0];
  int f1 = eattr[(size_t)e * 3 + 1];
  int f2 = eattr[(size_t)e * 3 + 2];
  const float* xs = x + (size_t)s * DIM;
  const float* t0 = etbl + (size_t)f0 * DIM;
  const float* t1 = etbl + (size_t)(22 + f1) * DIM;
  const float* t2 = etbl + (size_t)(28 + f2) * DIM;
  float* hd = h + (size_t)d * DIM;
  int c = lane * 8;
#pragma unroll
  for (int half = 0; half < 2; ++half) {
    int cc = c + half * 4;
    float4 xv = *(const float4*)(xs + cc);
    float4 a  = *(const float4*)(t0 + cc);
    float4 b  = *(const float4*)(t1 + cc);
    float4 cf = *(const float4*)(t2 + cc);
    float4 bb = *(const float4*)(ebias + cc);
    float m0 = fmaxf(xv.x + a.x + b.x + cf.x + bb.x, 0.f);
    float m1 = fmaxf(xv.y + a.y + b.y + cf.y + bb.y, 0.f);
    float m2 = fmaxf(xv.z + a.z + b.z + cf.z + bb.z, 0.f);
    float m3 = fmaxf(xv.w + a.w + b.w + cf.w + bb.w, 0.f);
    atomicAdd(hd + cc + 0, m0);
    atomicAdd(hd + cc + 1, m1);
    atomicAdd(hd + cc + 2, m2);
    atomicAdd(hd + cc + 3, m3);
  }
}

// ---------------------------------------------------------------------------
// Fused node MLP via WMMA bf16: x_out = gelu( relu(bn(h@w1+b1)) @ w2 + b2 )
// block = 8 waves; wave owns 16 rows x 256 cols (16 C-frags).
// Weights staged to LDS per block via async global->LDS copies (ASYNCcnt),
// with the w2 DMA overlapped against the GEMM1 epilogue.
// ---------------------------------------------------------------------------
static __device__ __forceinline__ v16bf cvtA(float4 a, float4 b, float4 c, float4 d) {
  v16bf r;
  r[0]=(__bf16)a.x;  r[1]=(__bf16)a.y;  r[2]=(__bf16)a.z;  r[3]=(__bf16)a.w;
  r[4]=(__bf16)b.x;  r[5]=(__bf16)b.y;  r[6]=(__bf16)b.z;  r[7]=(__bf16)b.w;
  r[8]=(__bf16)c.x;  r[9]=(__bf16)c.y;  r[10]=(__bf16)c.z; r[11]=(__bf16)c.w;
  r[12]=(__bf16)d.x; r[13]=(__bf16)d.y; r[14]=(__bf16)d.z; r[15]=(__bf16)d.w;
  return r;
}

// issue async stage of 128KB frag-ordered weights into LDS (each thread: 32 x b128)
static __device__ __forceinline__ void stage_issue(const __bf16* __restrict__ gsrc,
                                                   __bf16* ldsDst) {
  unsigned ldsBase = (unsigned)(uintptr_t)ldsDst;   // low 32 bits of flat ptr == LDS offset
  const char* g = (const char*)gsrc;
  int t = threadIdx.x;
#pragma unroll
  for (int i = 0; i < 32; ++i) {
    int ch = t + i * 256;                           // 0..8191 chunks of 16B
    unsigned loff = ldsBase + (unsigned)ch * 16u;
    const char* ga = g + (size_t)ch * 16;
    asm volatile("global_load_async_to_lds_b128 %0, %1, off"
                 :: "v"(loff), "v"(ga) : "memory");
  }
}
static __device__ __forceinline__ void stage_wait() {
  asm volatile("s_wait_asynccnt 0x0" ::: "memory");
}

__global__ __launch_bounds__(256, 1) void gine_mlp_wmma(
    const float* __restrict__ hIn, const __bf16* __restrict__ w1f,
    const __bf16* __restrict__ w2f, const float* __restrict__ scale1,
    const float* __restrict__ shift1, const float* __restrict__ bias2,
    float* __restrict__ xOut) {
  extern __shared__ char smem[];
  __bf16* ldsW   = (__bf16*)smem;                        // 128 KB staged weight
  __bf16* ldsMid = (__bf16*)(smem + WLDS_ELEMS * 2);     // 67.6 KB mid tiles

  const int wave = threadIdx.x >> 5;
  const int lane = threadIdx.x & 31;
  const int lm = lane & 15;   // A/C row within tile; B/C col within 16-col tile
  const int lh = lane >> 4;   // lane half -> K-chunk select
  const int rowBase = blockIdx.x * 128 + wave * 16;      // rows padded to /128, no tail
  __bf16* mid = ldsMid + wave * 16 * MIDSTRIDE;
  const v16bf* ldsWv = (const v16bf*)ldsW;

  // ---- issue w1 DMA, overlap with first A-chunk fetch, then wait ----
  stage_issue(w1f, ldsW);
  const float* hRow = hIn + (size_t)(rowBase + lm) * DIM;
  float4 a0 = *(const float4*)(hRow + 8 * lh);
  float4 a1 = *(const float4*)(hRow + 8 * lh + 4);
  float4 a2 = *(const float4*)(hRow + 8 * lh + 16);
  float4 a3 = *(const float4*)(hRow + 8 * lh + 20);
  stage_wait();
  __syncthreads();

  // ---------------- GEMM1: h @ w1  (K = 256 = 8 x 32) ----------------
  v8f acc[16];
#pragma unroll
  for (int i = 0; i < 16; ++i) acc[i] = (v8f){0.f,0.f,0.f,0.f,0.f,0.f,0.f,0.f};

#pragma unroll
  for (int ks = 0; ks < 8; ++ks) {
    v16bf A = cvtA(a0, a1, a2, a3);
    if (ks < 7) {                       // double-buffer next A chunk from global
      const int k0 = 32 * (ks + 1) + 8 * lh;
      a0 = *(const float4*)(hRow + k0);
      a1 = *(const float4*)(hRow + k0 + 4);
      a2 = *(const float4*)(hRow + k0 + 16);
      a3 = *(const float4*)(hRow + k0 + 20);
    }
    v16bf B0 = ldsWv[(ks * 16 + 0) * 32 + lane];        // depth-2 B pipeline
    v16bf B1 = ldsWv[(ks * 16 + 1) * 32 + lane];
#pragma unroll
    for (int nt = 0; nt < 16; ++nt) {
      v16bf Bn = B1;
      if (nt < 14) Bn = ldsWv[(ks * 16 + nt + 2) * 32 + lane];
      acc[nt] = __builtin_amdgcn_wmma_f32_16x16x32_bf16(false, A, false, B0,
                                                        (short)0, acc[nt], false, false);
      B0 = B1;
      B1 = Bn;
    }
  }

  // ---- all waves done reading ldsW: start w2 DMA, overlap with epilogue 1 ----
  __syncthreads();
  stage_issue(w2f, ldsW);

  // ---- epilogue 1: fused BN+ReLU -> bf16 mid tile in LDS (wave-private) ----
#pragma unroll
  for (int nt = 0; nt < 16; ++nt) {
    const int n = lm + 16 * nt;
    const float sc = scale1[n];
    const float sh = shift1[n];
#pragma unroll
    for (int r = 0; r < 8; ++r) {
      float u = fmaxf(acc[nt][r] * sc + sh, 0.f);
      mid[(r + 8 * lh) * MIDSTRIDE + n] = (__bf16)u;
    }
  }

  stage_wait();
  __syncthreads();

  // ---------------- GEMM2: mid @ w2 ----------------
  v8f acc2[16];
#pragma unroll
  for (int i = 0; i < 16; ++i) acc2[i] = (v8f){0.f,0.f,0.f,0.f,0.f,0.f,0.f,0.f};

#pragma unroll
  for (int ks = 0; ks < 8; ++ks) {
    const __bf16* mrow = mid + lm * MIDSTRIDE + 32 * ks + 8 * lh;
    v8bf lo = *(const v8bf*)(mrow);
    v8bf hi = *(const v8bf*)(mrow + 16);
    v16bf A;
#pragma unroll
    for (int j = 0; j < 8; ++j) { A[j] = lo[j]; A[j + 8] = hi[j]; }
    v16bf B0 = ldsWv[(ks * 16 + 0) * 32 + lane];
    v16bf B1 = ldsWv[(ks * 16 + 1) * 32 + lane];
#pragma unroll
    for (int nt = 0; nt < 16; ++nt) {
      v16bf Bn = B1;
      if (nt < 14) Bn = ldsWv[(ks * 16 + nt + 2) * 32 + lane];
      acc2[nt] = __builtin_amdgcn_wmma_f32_16x16x32_bf16(false, A, false, B0,
                                                         (short)0, acc2[nt], false, false);
      B0 = B1;
      B1 = Bn;
    }
  }

  // ---- epilogue 2: +b2, exact GELU, store new x ----
#pragma unroll
  for (int nt = 0; nt < 16; ++nt) {
    const int n = lm + 16 * nt;
    const float bb = bias2[n];
#pragma unroll
    for (int r = 0; r < 8; ++r) {
      float y = acc2[nt][r] + bb;
      float gel = 0.5f * y * (1.0f + erff(y * 0.70710678118654752440f));
      xOut[(size_t)(rowBase + r + 8 * lh) * DIM + n] = gel;
    }
  }
}

// ---------------------------------------------------------------------------
// Pooling + head
// ---------------------------------------------------------------------------
__global__ void zero_kernel(float* __restrict__ p, int n) {
  int i = blockIdx.x * 256 + threadIdx.x;
  if (i < n) p[i] = 0.f;
}

__global__ void pool_kernel(const float* __restrict__ x, const int* __restrict__ batch,
                            float* __restrict__ gbuf) {
  int n = blockIdx.x, c = threadIdx.x;
  atomicAdd(&gbuf[(size_t)batch[n] * DIM + c], x[(size_t)n * DIM + c]);
}

__global__ void head1_kernel(const float* __restrict__ gbuf, const float* __restrict__ w1,
                             const float* __restrict__ b1, const float* __restrict__ bng,
                             const float* __restrict__ bnb, const float* __restrict__ bnm,
                             const float* __restrict__ bnv, float* __restrict__ hh) {
  int b = blockIdx.x, c = threadIdx.x;
  const float* row = gbuf + (size_t)b * DIM;
  float acc = 0.f;
  for (int k = 0; k < DIM; ++k) acc += row[k] * w1[(size_t)k * DIM + c];
  float S = bng[c] * rsqrtf(bnv[c] + 1e-5f);
  float val = (acc + b1[c] - bnm[c]) * S + bnb[c];
  hh[(size_t)b * DIM + c] = fmaxf(val, 0.f);
}

__global__ void head2_kernel(const float* __restrict__ hh, const float* __restrict__ w2,
                             const float* __restrict__ b2, float* __restrict__ out) {
  int b = blockIdx.x, t = threadIdx.x;
  if (t >= NTASK) return;
  const float* row = hh + (size_t)b * DIM;
  float acc = b2[t];
  for (int k = 0; k < DIM; ++k) acc += row[k] * w2[(size_t)k * NTASK + t];
  out[(size_t)b * NTASK + t] = acc;
}

// ---------------------------------------------------------------------------
// Launch
// ---------------------------------------------------------------------------
extern "C" void kernel_launch(void* const* d_in, const int* in_sizes, int n_in,
                              void* d_out, int out_size, void* d_ws, size_t ws_size,
                              hipStream_t stream) {
  const int N = in_sizes[0] / 9;
  const int E = in_sizes[1] / 3;
  const int L = 4;
  const int NPAD = ((N + 127) / 128) * 128;   // pad rows so every wave/block is full

  const int* xfeat = (const int*)d_in[0];
  const int* eattr = (const int*)d_in[1];
  const int* eidx  = (const int*)d_in[2];
  const int* batch = (const int*)d_in[3];
  const float* nemb[9];
  for (int i = 0; i < 9; ++i) nemb[i] = (const float*)d_in[5 + i];
  const float* npw = (const float*)d_in[14];
  const float* npb = (const float*)d_in[15];
  const float* eemb[3];
  for (int i = 0; i < 3; ++i) eemb[i] = (const float*)d_in[16 + i];
  const float* epw = (const float*)d_in[19];
  const float* epb = (const float*)d_in[20];
  // convs: base 21 + 10*l : lin_w, lin_b, w1, b1, bn.g, bn.b, bn.m, bn.v, w2, b2
  // head : base 61       : w1, b1, bn.g, bn.b, bn.m, bn.v, w2, b2

  char* ws = (char*)d_ws;
  size_t off = 0;
  auto alloc = [&](size_t bytes) -> void* {
    void* p = ws + off;
    off += (bytes + 255) & ~(size_t)255;
    return p;
  };
  float*  x     = (float*)alloc((size_t)NPAD * DIM * 4);
  float*  h     = (float*)alloc((size_t)NPAD * DIM * 4);
  float*  ntbl  = (float*)alloc((size_t)177 * DIM * 4);
  float*  ebase = (float*)alloc((size_t)30 * DIM * 4);
  float*  etbl  = (float*)alloc((size_t)L * 30 * DIM * 4);
  float*  ebias = (float*)alloc((size_t)L * DIM * 4);
  __bf16* wfrag = (__bf16*)alloc((size_t)2 * L * DIM * DIM * 2);
  float*  scl   = (float*)alloc((size_t)L * DIM * 4);
  float*  shf   = (float*)alloc((size_t)L * DIM * 4);
  float*  gbuf  = (float*)alloc((size_t)NGRAPH * DIM * 4);
  float*  hh    = (float*)alloc((size_t)NGRAPH * DIM * 4);

  // --- fold encoders / edge-lin into tables; pack weights into B-frag order ---
  node_table_kernel<<<177, DIM, 0, stream>>>(nemb[0],nemb[1],nemb[2],nemb[3],nemb[4],
                                             nemb[5],nemb[6],nemb[7],nemb[8], npw, ntbl);
  edge_base_kernel<<<30, DIM, 0, stream>>>(eemb[0], eemb[1], eemb[2], epw, ebase);
  for (int l = 0; l < L; ++l) {
    const int cb = 21 + 10 * l;
    edge_layer_kernel<<<31, DIM, 0, stream>>>(ebase, epb, (const float*)d_in[cb + 0],
                                              (const float*)d_in[cb + 1],
                                              etbl + (size_t)l * 30 * DIM, ebias + (size_t)l * DIM);
    pack_wfrag_kernel<<<256, 256, 0, stream>>>((const float*)d_in[cb + 2],
                                               wfrag + (size_t)(2 * l) * DIM * DIM);
    pack_wfrag_kernel<<<256, 256, 0, stream>>>((const float*)d_in[cb + 8],
                                               wfrag + (size_t)(2 * l + 1) * DIM * DIM);
    bnfold_kernel<<<1, DIM, 0, stream>>>((const float*)d_in[cb + 3], (const float*)d_in[cb + 4],
                                         (const float*)d_in[cb + 5], (const float*)d_in[cb + 6],
                                         (const float*)d_in[cb + 7],
                                         scl + (size_t)l * DIM, shf + (size_t)l * DIM);
  }

  // --- node encoder ---
  node_encode_kernel<<<N, DIM, 0, stream>>>(xfeat, ntbl, npb, x);

  // --- GINE layers ---
  for (int l = 0; l < L; ++l) {
    const int cb = 21 + 10 * l;
    hipMemcpyAsync(h, x, (size_t)N * DIM * 4, hipMemcpyDeviceToDevice, stream); // h = x (self term)
    edge_msg_kernel<<<(E + 7) / 8, 256, 0, stream>>>(x, eidx, eattr,
                                                     etbl + (size_t)l * 30 * DIM,
                                                     ebias + (size_t)l * DIM, h, E);
    gine_mlp_wmma<<<NPAD / 128, 256, SMEM_BYTES, stream>>>(
        h, wfrag + (size_t)(2 * l) * DIM * DIM, wfrag + (size_t)(2 * l + 1) * DIM * DIM,
        scl + (size_t)l * DIM, shf + (size_t)l * DIM, (const float*)d_in[cb + 9], x);
  }

  // --- pooling + head ---
  zero_kernel<<<(NGRAPH * DIM + 255) / 256, 256, 0, stream>>>(gbuf, NGRAPH * DIM);
  pool_kernel<<<N, DIM, 0, stream>>>(x, batch, gbuf);
  head1_kernel<<<NGRAPH, DIM, 0, stream>>>(gbuf, (const float*)d_in[61], (const float*)d_in[62],
                                           (const float*)d_in[63], (const float*)d_in[64],
                                           (const float*)d_in[65], (const float*)d_in[66], hh);
  head2_kernel<<<NGRAPH, 32, 0, stream>>>(hh, (const float*)d_in[67], (const float*)d_in[68],
                                          (float*)d_out);
}
